// RNN_69526930588180
// MI455X (gfx1250) — compile-verified, matrix operand order checked by reading, not run on hardware
//
#include <hip/hip_runtime.h>
#include <hip/hip_bf16.h>
#include <math.h>

// Problem constants (match reference).
constexpr int Bc = 256;   // batch
constexpr int Tc = 256;   // seq len
constexpr int Dc = 64;    // input dim
constexpr int Hc = 512;   // hidden dim

typedef __attribute__((ext_vector_type(16))) __bf16 v16bf;
typedef __attribute__((ext_vector_type(8)))  __bf16 v8bf;
typedef __attribute__((ext_vector_type(8)))  float  v8f;

// Assemble a 16-element bf16 WMMA fragment from two contiguous 16-byte chunks.
// Layout per CDNA5 ISA (16-bit A/B 16x32): lane = row/col (lane&15), half = lane>>4;
// elements [0..7] = K in [half*8, half*8+8), elements [8..15] = K in [16+half*8, ...).
__device__ __forceinline__ v16bf load_frag(const __bf16* p0, const __bf16* p1) {
    v8bf lo = *(const v8bf*)p0;
    v8bf hi = *(const v8bf*)p1;
    v16bf r;
#pragma unroll
    for (int i = 0; i < 8; ++i) { r[i] = lo[i]; r[i + 8] = hi[i]; }
    return r;
}

// ---------------------------------------------------------------------------
// Prep kernels: fp32 -> bf16 conversions (weights transposed to [N][K]).
// ---------------------------------------------------------------------------
__global__ void k_cvt_bf16(const float* __restrict__ in, __bf16* __restrict__ out, int n) {
    int i = blockIdx.x * 256 + threadIdx.x;
    if (i < n) out[i] = (__bf16)in[i];
}

__global__ void k_transpose_bf16(const float* __restrict__ W, __bf16* __restrict__ WT,
                                 int K, int N) {
    int i = blockIdx.x * 256 + threadIdx.x;
    if (i < K * N) {
        int k = i / N, n = i - k * N;
        WT[(size_t)n * K + k] = (__bf16)W[i];   // WT[n][k] = W[k][n]
    }
}

// ---------------------------------------------------------------------------
// Projection GEMM: XP[t][b][n] = sum_k A[b*T+t][k] * W[k][n] + bias[n]
// A: bf16 [M=B*T][K] row-major; WT: bf16 [N=H][K]; XP: fp32 [T][B][H].
// Block = 256 threads = 8 waves; wave tile = 16 rows x 64 cols (4 WMMA n-tiles);
// workgroup tile = 128 rows x 64 cols.
// ---------------------------------------------------------------------------
__global__ void __launch_bounds__(256)
k_proj(const __bf16* __restrict__ A, const __bf16* __restrict__ WT,
       const float* __restrict__ bias, float* __restrict__ XP, int K) {
    const int lane = threadIdx.x & 31;
    const int wave = threadIdx.x >> 5;
    const int lrow = lane & 15;
    const int half = lane >> 4;
    const int mbase = blockIdx.x * 128 + wave * 16;
    const int nbase = blockIdx.y * 64;

    v8f acc[4] = {};
    const __bf16* arow = A + (size_t)(mbase + lrow) * K;

    for (int kc = 0; kc < K; kc += 32) {
        v16bf af = load_frag(arow + kc + half * 8, arow + kc + 16 + half * 8);
#pragma unroll
        for (int nt = 0; nt < 4; ++nt) {
            const __bf16* brow = WT + (size_t)(nbase + nt * 16 + lrow) * K;
            v16bf bf = load_frag(brow + kc + half * 8, brow + kc + 16 + half * 8);
            acc[nt] = __builtin_amdgcn_wmma_f32_16x16x32_bf16(
                false, af, false, bf, (short)0, acc[nt], false, false);
        }
    }

#pragma unroll
    for (int nt = 0; nt < 4; ++nt) {
        const int n = nbase + nt * 16 + lrow;
        const float bv = bias[n];
#pragma unroll
        for (int r = 0; r < 8; ++r) {
            const int m = mbase + r + 8 * half;   // m = b*T + t
            const int b = m / Tc, t = m % Tc;
            XP[((size_t)t * Bc + b) * Hc + n] = acc[nt][r] + bv;
        }
    }
}

// ---------------------------------------------------------------------------
// Recurrence: h_t = tanh(XP_t + h_{t-1} @ Wh), batch-partitioned (no inter-WG
// sync needed). Each workgroup owns 16 batch rows; h state lives in LDS (bf16).
// Block = 256 threads = 8 waves; wave handles 4 n-tiles (64 columns).
// WhT: bf16 [H][H] = [n][k] so B-fragments are contiguous loads.
// ---------------------------------------------------------------------------
__global__ void __launch_bounds__(256)
k_recurrent(const float* __restrict__ XP,      // [T][B][H] fp32
            const __bf16* __restrict__ WhT,    // [H][H] bf16 transposed
            __bf16* __restrict__ hseq,         // [B][T][H] bf16 (null if !store_seq)
            float* __restrict__ hlast,         // [B][H] fp32 (may be null)
            int store_seq) {
    __shared__ __bf16 hcur[16][Hc + 8];        // +8 bf16 pad -> 4-bank row skew

    const int lane = threadIdx.x & 31;
    const int wave = threadIdx.x >> 5;
    const int lrow = lane & 15;
    const int half = lane >> 4;
    const int b0 = blockIdx.x * 16;
    const int ncol0 = wave * 64;

    // h_0 = 0
    for (int i = threadIdx.x; i < 16 * (Hc + 8); i += 256)
        (&hcur[0][0])[i] = (__bf16)0.0f;
    __syncthreads();

    for (int t = 0; t < Tc; ++t) {
        const float* xpbase = XP + ((size_t)t * Bc + b0) * Hc;

        // Prefetch next step's XP tile (32 KB) while this step computes.
        if (t + 1 < Tc)
            __builtin_prefetch(xpbase + (size_t)Bc * Hc + threadIdx.x * 32, 0, 0);

        // Accumulators initialized with the precomputed input projection.
        v8f acc[4];
#pragma unroll
        for (int nt = 0; nt < 4; ++nt) {
            const int n = ncol0 + nt * 16 + lrow;
#pragma unroll
            for (int r = 0; r < 8; ++r)
                acc[nt][r] = xpbase[(size_t)(r + 8 * half) * Hc + n];
        }

        // h_{t-1} @ Wh via WMMA: A from LDS, B streamed from L2-resident WhT.
#pragma unroll 2
        for (int kc = 0; kc < Hc; kc += 32) {
            v16bf af = load_frag(&hcur[lrow][kc + half * 8],
                                 &hcur[lrow][kc + 16 + half * 8]);
#pragma unroll
            for (int nt = 0; nt < 4; ++nt) {
                const __bf16* brow = WhT + (size_t)(ncol0 + nt * 16 + lrow) * Hc;
                v16bf bf = load_frag(brow + kc + half * 8, brow + kc + 16 + half * 8);
                acc[nt] = __builtin_amdgcn_wmma_f32_16x16x32_bf16(
                    false, af, false, bf, (short)0, acc[nt], false, false);
            }
        }

        __syncthreads();   // everyone done reading h_{t-1}

#pragma unroll
        for (int nt = 0; nt < 4; ++nt) {
            const int n = ncol0 + nt * 16 + lrow;
#pragma unroll
            for (int r = 0; r < 8; ++r) {
                const int m = r + 8 * half;
                const float v = tanhf(acc[nt][r]);
                const __bf16 hv = (__bf16)v;
                hcur[m][n] = hv;
                if (store_seq)
                    hseq[((size_t)(b0 + m) * Tc + t) * Hc + n] = hv;
                if (hlast && t == Tc - 1)
                    hlast[(size_t)(b0 + m) * Hc + n] = v;
            }
        }
        __syncthreads();   // h_t visible before next step reads it
    }
}

// ---------------------------------------------------------------------------
// Final dense: out[b] = hlast[b,:] . Wf + bf   (B=256 rows, trivial)
// ---------------------------------------------------------------------------
__global__ void k_final(const float* __restrict__ hl, const float* __restrict__ Wf,
                        const float* __restrict__ bfv, float* __restrict__ out) {
    int b = blockIdx.x * 256 + threadIdx.x;
    if (b < Bc) {
        float s = bfv[0];
        for (int k = 0; k < Hc; ++k) s += hl[(size_t)b * Hc + k] * Wf[k];
        out[b] = s;
    }
}

// ---------------------------------------------------------------------------
// Launch
// ---------------------------------------------------------------------------
extern "C" void kernel_launch(void* const* d_in, const int* in_sizes, int n_in,
                              void* d_out, int out_size, void* d_ws, size_t ws_size,
                              hipStream_t stream) {
    const float* x   = (const float*)d_in[0];
    const float* Wx0 = (const float*)d_in[1];
    const float* Wh0 = (const float*)d_in[2];
    const float* b0  = (const float*)d_in[3];
    const float* Wx1 = (const float*)d_in[4];
    const float* Wh1 = (const float*)d_in[5];
    const float* b1  = (const float*)d_in[6];
    const float* Wx2 = (const float*)d_in[7];
    const float* Wh2 = (const float*)d_in[8];
    const float* b2  = (const float*)d_in[9];
    const float* Wf  = (const float*)d_in[10];
    const float* bfv = (const float*)d_in[11];

    // Workspace layout (all offsets 256B-aligned), ~203 MB total.
    char* ws = (char*)d_ws;
    const size_t M = (size_t)Bc * Tc;                 // 65536 rows
    __bf16* xbf   = (__bf16*)(ws + 0);                //  8 MB : x as bf16 [M][D]
    __bf16* hseq  = (__bf16*)(ws + 8388608);          // 64 MB : [B][T][H] bf16
    float*  XP    = (float*)(ws + 75497472);          //128 MB : [T][B][H] fp32
    __bf16* wx0t  = (__bf16*)(ws + 209715200);        // [H][D]
    __bf16* wh0t  = (__bf16*)(ws + 209780736);        // [H][H]
    __bf16* wx1t  = (__bf16*)(ws + 210305024);
    __bf16* wh1t  = (__bf16*)(ws + 210829312);
    __bf16* wx2t  = (__bf16*)(ws + 211353600);
    __bf16* wh2t  = (__bf16*)(ws + 211877888);
    float*  hlast = (float*)(ws + 212402176);         // [B][H] fp32

    // --- weight / input conversion ---
    k_cvt_bf16<<<(int)((M * Dc + 255) / 256), 256, 0, stream>>>(x, xbf, (int)(M * Dc));
    k_transpose_bf16<<<(Dc * Hc + 255) / 256, 256, 0, stream>>>(Wx0, wx0t, Dc, Hc);
    k_transpose_bf16<<<(Hc * Hc + 255) / 256, 256, 0, stream>>>(Wh0, wh0t, Hc, Hc);
    k_transpose_bf16<<<(Hc * Hc + 255) / 256, 256, 0, stream>>>(Wx1, wx1t, Hc, Hc);
    k_transpose_bf16<<<(Hc * Hc + 255) / 256, 256, 0, stream>>>(Wh1, wh1t, Hc, Hc);
    k_transpose_bf16<<<(Hc * Hc + 255) / 256, 256, 0, stream>>>(Wx2, wx2t, Hc, Hc);
    k_transpose_bf16<<<(Hc * Hc + 255) / 256, 256, 0, stream>>>(Wh2, wh2t, Hc, Hc);

    const dim3 projGrid((unsigned)(M / 128), Hc / 64);

    // --- layer 0 ---
    k_proj<<<projGrid, 256, 0, stream>>>(xbf, wx0t, b0, XP, Dc);
    k_recurrent<<<Bc / 16, 256, 0, stream>>>(XP, wh0t, hseq, nullptr, 1);
    // --- layer 1 (hseq fully consumed by k_proj before being overwritten) ---
    k_proj<<<projGrid, 256, 0, stream>>>(hseq, wx1t, b1, XP, Hc);
    k_recurrent<<<Bc / 16, 256, 0, stream>>>(XP, wh1t, hseq, nullptr, 1);
    // --- layer 2 (final state only) ---
    k_proj<<<projGrid, 256, 0, stream>>>(hseq, wx2t, b2, XP, Hc);
    k_recurrent<<<Bc / 16, 256, 0, stream>>>(XP, wh2t, nullptr, hlast, 0);
    // --- final dense ---
    k_final<<<1, 256, 0, stream>>>(hlast, Wf, bfv, (float*)d_out);
}
